// LigerAttention_39651138076964
// MI455X (gfx1250) — matrix operand order repaired and checked by reading, not compiled, not run on hardware
//
#include <hip/hip_runtime.h>
#include <math.h>

// ---------------------------------------------------------------------------
// LigerAttention for MI455X (gfx1250, wave32, WMMA).
// Shapes: B=1, L=4096, HID=2048, H=32, KVH=8, D=64, WIN=64 (chunk=64).
// ---------------------------------------------------------------------------

typedef _Float16 f16;
typedef __attribute__((ext_vector_type(16))) _Float16 v16h;
typedef __attribute__((ext_vector_type(8)))  _Float16 v8h;
typedef __attribute__((ext_vector_type(8)))  float    v8f;
typedef __attribute__((ext_vector_type(4)))  float    v4f;

#define SEQ_L   4096
#define HID_E   2048
#define NHEAD   32
#define NKVH    8
#define HDIM    64
#define KVW     (NKVH * HDIM)   // 512
#define WIN_C   64
#define NCHUNK  (SEQ_L / WIN_C) // 64
#define LN_THETA 9.210340371976184f   // ln(10000)

// Build a 16-half fragment from two contiguous 8-half runs in LDS
// (each run -> one ds_load_b128).
__device__ __forceinline__ v16h lds_frag16(const f16* p0, const f16* p1)
{
    v8h lo = *(const v8h*)p0;
    v8h hi = *(const v8h*)p1;
    return __builtin_shufflevector(lo, hi,
        0, 1, 2, 3, 4, 5, 6, 7, 8, 9, 10, 11, 12, 13, 14, 15);
}

// ===========================================================================
// Generic f16-WMMA GEMM:  C[M,N] = cvt_f16(A) @ cvt_f16(W),  all row-major f32.
// N is a template constant so the column-strided staging reads become
// independent global_load_b32 with immediate offsets (no 64-bit addr chain).
// If A2 != null, effective A is 0.5*(A + A2) (used for the final Wo GEMM).
// Block = 128 threads (4 waves). Block tile 64x64; wave w owns rows +16*w.
// W tile is staged in LDS in *fragment-packed* layout (slot == consuming
// lane's 16 halves) with double buffering: stage k+1 while WMMAs eat k.
// ===========================================================================
template<int N>
__global__ __launch_bounds__(128)
void wmma_gemm_kernel(const float* __restrict__ A, const float* __restrict__ A2,
                      const float* __restrict__ W, float* __restrict__ C, int K)
{
    __shared__ __align__(32) f16 lds_bf[2][128 * 16];   // 2 x 4 KB, frag-packed

    const int tid   = threadIdx.x;
    const int wave  = tid >> 5;
    const int lane  = tid & 31;
    const int lhalf = lane >> 4;   // 0 or 1
    const int l15   = lane & 15;

    const int colBase = blockIdx.x * 64;
    const int rowBase = blockIdx.y * 64 + wave * 16;

    // staging identity: thread tid fills frag slot (nt = tid>>5, lane2 = tid&31)
    const float* Wst = W + (size_t)(tid & 16) * N          // K offset 0 or 16
                         + (colBase + ((tid >> 5) << 4) + (tid & 15));

    auto stage = [&](int buf, int k0) {
        const float* p = Wst + (size_t)k0 * N;
        v16h wf;
        #pragma unroll
        for (int j = 0; j < 16; ++j)
            wf[j] = (f16)p[j * N];                          // immediate offsets
        *(v16h*)&lds_bf[buf][tid * 16] = wf;
        __builtin_prefetch(p + (size_t)64 * N, 0, 0);       // 2 tiles ahead
    };

    v8f acc[4];
    #pragma unroll
    for (int nt = 0; nt < 4; ++nt)
        #pragma unroll
        for (int i = 0; i < 8; ++i) acc[nt][i] = 0.0f;

    const float* Arow  = A + (size_t)(rowBase + l15) * K;
    const float* A2row = A2 ? (A2 + (size_t)(rowBase + l15) * K) : nullptr;

    stage(0, 0);
    int buf = 0;
    for (int k0 = 0; k0 < K; k0 += 32) {
        __syncthreads();                 // staging of 'buf' complete,
                                         // readers of buf^1 (prev iter) done
        if (k0 + 32 < K) stage(buf ^ 1, k0 + 32);

        // --- A fragment: two contiguous 8-float runs -> 4x b128 loads ---
        v16h a;
        {
            const int kb0 = k0 + (lhalf << 4);
            const int kb1 = kb0 + 8;
            v4f x[4];
            x[0] = *(const v4f*)(Arow + kb0);
            x[1] = *(const v4f*)(Arow + kb0 + 4);
            x[2] = *(const v4f*)(Arow + kb1);
            x[3] = *(const v4f*)(Arow + kb1 + 4);
            if (A2row) {
                v4f y0 = *(const v4f*)(A2row + kb0);
                v4f y1 = *(const v4f*)(A2row + kb0 + 4);
                v4f y2 = *(const v4f*)(A2row + kb1);
                v4f y3 = *(const v4f*)(A2row + kb1 + 4);
                x[0] = 0.5f * (x[0] + y0);
                x[1] = 0.5f * (x[1] + y1);
                x[2] = 0.5f * (x[2] + y2);
                x[3] = 0.5f * (x[3] + y3);
            }
            #pragma unroll
            for (int j = 0; j < 16; ++j)
                a[j] = (f16)x[j >> 2][j & 3];
        }

        // --- 4 WMMAs reusing the A fragment (16x64 per wave) ---
        #pragma unroll
        for (int nt = 0; nt < 4; ++nt) {
            v16h b = *(const v16h*)&lds_bf[buf][((nt << 5) + lane) * 16];
            acc[nt] = __builtin_amdgcn_wmma_f32_16x16x32_f16(
                false, a, false, b, (short)0, acc[nt], false, false);
        }
        buf ^= 1;
    }

    // --- store C per 32-bit C/D 16x16 layout (immediate row offsets) ---
    float* Crow = C + (size_t)(rowBase + (lhalf << 3)) * N;
    #pragma unroll
    for (int nt = 0; nt < 4; ++nt) {
        int n = colBase + (nt << 4) + l15;
        #pragma unroll
        for (int r = 0; r < 8; ++r)
            Crow[(size_t)r * N + n] = acc[nt][r];
    }
}

// ===========================================================================
// Gate: gl = logsigmoid(k) / 16   (pool ratio == 1 -> g == k_flat)
// ===========================================================================
__global__ __launch_bounds__(256)
void gate_kernel(const float* __restrict__ k, float* __restrict__ gl, int n)
{
    int i = blockIdx.x * 256 + threadIdx.x;
    if (i < n) {
        float x  = k[i];
        float ls = fminf(x, 0.0f) - log1pf(__expf(-fabsf(x)));
        gl[i] = ls * (1.0f / 16.0f);
    }
}

// ===========================================================================
// In-place softmax over rows of width 64 (one wave32 per row, 2 elems/lane).
// ===========================================================================
__global__ __launch_bounds__(256)
void softmax64_kernel(float* __restrict__ x, int nrows)
{
    int row  = blockIdx.x * 8 + (threadIdx.x >> 5);
    int lane = threadIdx.x & 31;
    if (row >= nrows) return;
    float* p = x + (size_t)row * 64;
    float a = p[lane], b = p[lane + 32];
    float m = fmaxf(a, b);
    #pragma unroll
    for (int s = 16; s >= 1; s >>= 1) m = fmaxf(m, __shfl_xor(m, s, 32));
    float ea = __expf(a - m), eb = __expf(b - m);
    float sum = ea + eb;
    #pragma unroll
    for (int s = 16; s >= 1; s >>= 1) sum += __shfl_xor(sum, s, 32);
    float inv = 1.0f / sum;
    p[lane]      = ea * inv;
    p[lane + 32] = eb * inv;
}

// ===========================================================================
// Sliding-window attention branch (RoPE + banded causal softmax), WMMA tiles.
// One wave per (head h, chunk c). Scores: q(64x64) @ k^T(64x128). PV: P @ V.
// LDS layouts chosen so every fragment is a contiguous run of halves:
//   qs[m][d]  row-major  -> A fragments   = 2x v8h
//   ks[j][d]  row-major  -> score B frags = contiguous 16 halves
//   vsT[d][j] TRANSPOSED -> PV B frags    = contiguous 16 halves
//   ps[m][n]  row-major  -> PV A frags    = 2x v8h
// ===========================================================================
__global__ __launch_bounds__(32)
void swa_kernel(const float* __restrict__ q, const float* __restrict__ k,
                const float* __restrict__ v, float* __restrict__ y)
{
    const int h  = blockIdx.x;      // 0..31
    const int c  = blockIdx.y;      // 0..63
    const int hk = h >> 2;          // GQA: 4 q-heads per kv-head
    const int lane  = threadIdx.x;
    const int lhalf = lane >> 4;
    const int l15   = lane & 15;

    __shared__ __align__(32) f16 qs[64 * 64];     // RoPE'd q chunk
    __shared__ __align__(32) f16 ks[128 * 64];    // RoPE'd [prev|self] key rows
    __shared__ __align__(32) f16 vsT[64 * 128];   // values, transposed (d, j)
    __shared__ __align__(32) f16 ps[64 * 128];    // softmax probabilities

    // ---- stage q with RoPE ----
    for (int t = lane; t < 64 * 64; t += 32) {
        int m = t >> 6, d = t & 63;
        int l = c * WIN_C + m;
        float ang = (float)l * __expf(-(float)(d & 31) * (LN_THETA / 32.0f));
        float sn, cs; __sincosf(ang, &sn, &cs);
        const float* qr = q + (size_t)l * HID_E + h * HDIM;
        float x   = qr[d];
        float rot = (d < 32) ? -qr[d + 32] : qr[d - 32];
        qs[t] = (f16)(x * cs + rot * sn);
    }
    // ---- stage k (RoPE) and v^T, rows 0..63 = prev chunk, 64..127 = self ----
    for (int t = lane; t < 128 * 64; t += 32) {
        int j = t >> 6, d = t & 63;
        float kk = 0.0f, vv = 0.0f;
        if (j >= 64 || c > 0) {
            int l = (j < 64) ? (c - 1) * WIN_C + j : c * WIN_C + (j - 64);
            const float* kr = k + (size_t)l * KVW + hk * HDIM;
            float ang = (float)l * __expf(-(float)(d & 31) * (LN_THETA / 32.0f));
            float sn, cs; __sincosf(ang, &sn, &cs);
            float x   = kr[d];
            float rot = (d < 32) ? -kr[d + 32] : kr[d - 32];
            kk = x * cs + rot * sn;
            vv = v[(size_t)l * KVW + hk * HDIM + d];
        }
        ks[t]            = (f16)kk;
        vsT[d * 128 + j] = (f16)vv;
    }
    __syncthreads();

    const float scale = 0.125f;  // 1/sqrt(64)

    for (int mt = 0; mt < 4; ++mt) {
        // A fragments (two K-steps of 32 over d) for rows mt*16..mt*16+15
        const int am = (mt * 16 + l15) * 64;
        v16h afrag[2];
        #pragma unroll
        for (int ks0 = 0; ks0 < 2; ++ks0) {
            const f16* p0 = &qs[am + ks0 * 32 + (lhalf << 4)];
            afrag[ks0] = lds_frag16(p0, p0 + 8);
        }
        // ---- scores: 8 N-tiles of 16 key rows ----
        float sc[8][8];
        #pragma unroll
        for (int nt = 0; nt < 8; ++nt) {
            v8f accv;
            #pragma unroll
            for (int i = 0; i < 8; ++i) accv[i] = 0.0f;
            const int bn = (nt * 16 + l15) * 64;       // key row base
            #pragma unroll
            for (int ks0 = 0; ks0 < 2; ++ks0) {
                const f16* p0 = &ks[bn + ks0 * 32 + (lhalf << 4)];
                v16h b = lds_frag16(p0, p0 + 8);
                accv = __builtin_amdgcn_wmma_f32_16x16x32_f16(
                    false, afrag[ks0], false, b, (short)0, accv, false, false);
            }
            #pragma unroll
            for (int r = 0; r < 8; ++r) sc[nt][r] = accv[r];
        }
        // ---- mask + row softmax in C-layout registers ----
        #pragma unroll
        for (int r = 0; r < 8; ++r) {
            int mrow = mt * 16 + r + (lhalf << 3);  // query row within chunk
            float mx = -3.0e38f;
            #pragma unroll
            for (int nt = 0; nt < 8; ++nt) {
                int nabs = nt * 16 + l15;
                bool valid = (nabs < 64) ? ((c > 0) && (nabs >= mrow))
                                         : ((nabs - 64) <= mrow);
                sc[nt][r] = valid ? sc[nt][r] * scale : -1.0e30f;
                mx = fmaxf(mx, sc[nt][r]);
            }
            #pragma unroll
            for (int s = 8; s >= 1; s >>= 1) mx = fmaxf(mx, __shfl_xor(mx, s, 32));
            float sum = 0.0f;
            #pragma unroll
            for (int nt = 0; nt < 8; ++nt) { sc[nt][r] = __expf(sc[nt][r] - mx); sum += sc[nt][r]; }
            #pragma unroll
            for (int s = 8; s >= 1; s >>= 1) sum += __shfl_xor(sum, s, 32);
            float inv = 1.0f / sum;
            #pragma unroll
            for (int nt = 0; nt < 8; ++nt) sc[nt][r] *= inv;
        }
        // ---- P -> LDS (re-layout accumulator -> A fragment) ----
        #pragma unroll
        for (int nt = 0; nt < 8; ++nt)
            #pragma unroll
            for (int r = 0; r < 8; ++r) {
                int mrow = mt * 16 + r + (lhalf << 3);
                ps[mrow * 128 + nt * 16 + l15] = (f16)sc[nt][r];
            }
        __syncthreads();
        // ---- PV: out(16x64) for this row tile, K=128 in 4 steps ----
        const int pm = (mt * 16 + l15) * 128;
        #pragma unroll
        for (int vt = 0; vt < 4; ++vt) {
            v8f accv;
            #pragma unroll
            for (int i = 0; i < 8; ++i) accv[i] = 0.0f;
            const int bv = (vt * 16 + l15) * 128;      // vsT row base (dv)
            #pragma unroll
            for (int kk = 0; kk < 4; ++kk) {
                const f16* pa = &ps[pm + kk * 32 + (lhalf << 4)];
                v16h a = lds_frag16(pa, pa + 8);
                const f16* pb = &vsT[bv + kk * 32 + (lhalf << 4)];
                v16h b = lds_frag16(pb, pb + 8);
                accv = __builtin_amdgcn_wmma_f32_16x16x32_f16(
                    false, a, false, b, (short)0, accv, false, false);
            }
            #pragma unroll
            for (int r = 0; r < 8; ++r) {
                int mrow = mt * 16 + r + (lhalf << 3);
                int l = c * WIN_C + mrow;
                y[(size_t)l * HID_E + h * HDIM + vt * 16 + l15] = accv[r];
            }
        }
        __syncthreads();
    }
}

// ===========================================================================
// GLA recurrent scan. One block per head (32 blocks), 64 threads; thread owns
// one V-column of the 64x64 state in registers. Sequential over L=4096.
// ===========================================================================
__global__ __launch_bounds__(64)
void gla_kernel(const float* __restrict__ ql, const float* __restrict__ kl,
                const float* __restrict__ gl, const float* __restrict__ v,
                float* __restrict__ o_li)
{
    const int h  = blockIdx.x;
    const int hk = h >> 2;
    const int vd = threadIdx.x;
    __shared__ __align__(16) float sq[64], sk[64], seg[64], sv[64];

    float S[64];
    #pragma unroll
    for (int i = 0; i < 64; ++i) S[i] = 0.0f;

    for (int l = 0; l < SEQ_L; ++l) {
        sq[vd]  = ql[(size_t)l * HID_E + h * HDIM + vd];
        sk[vd]  = kl[(size_t)l * KVW + hk * HDIM + vd];
        seg[vd] = __expf(gl[(size_t)l * KVW + hk * HDIM + vd]);
        sv[vd]  = v[(size_t)l * KVW + hk * HDIM + vd];
        __syncthreads();
        float vv  = sv[vd];
        float acc = 0.0f;
        #pragma unroll
        for (int kd = 0; kd < 64; ++kd) {
            S[kd] = S[kd] * seg[kd] + sk[kd] * vv;
            acc   = fmaf(sq[kd], S[kd], acc);
        }
        o_li[(size_t)l * HID_E + h * HDIM + vd] = acc;
        __syncthreads();
    }
}

// ===========================================================================
// Host-side launch. Inputs: hidden_states, Wq, Wk, Wv, Wo (all f32).
// Output: f32 [4096, 2048]. Workspace layout (floats):
//   q[4096*2048] | k[4096*512] | v[4096*512] | gl[4096*512]
//   | y[4096*2048] | o_li[4096*2048]     (~126 MB total)
// ===========================================================================
extern "C" void kernel_launch(void* const* d_in, const int* in_sizes, int n_in,
                              void* d_out, int out_size, void* d_ws, size_t ws_size,
                              hipStream_t stream)
{
    (void)in_sizes; (void)n_in; (void)out_size; (void)ws_size;
    const float* hs = (const float*)d_in[0];
    const float* Wq = (const float*)d_in[1];
    const float* Wk = (const float*)d_in[2];
    const float* Wv = (const float*)d_in[3];
    const float* Wo = (const float*)d_in[4];
    float* out = (float*)d_out;

    float* ws   = (float*)d_ws;
    float* qb   = ws;                                   // 4096*2048
    float* kb   = qb + (size_t)SEQ_L * HID_E;           // 4096*512
    float* vb   = kb + (size_t)SEQ_L * KVW;             // 4096*512
    float* glb  = vb + (size_t)SEQ_L * KVW;             // 4096*512
    float* yb   = glb + (size_t)SEQ_L * KVW;            // 4096*2048
    float* olib = yb + (size_t)SEQ_L * HID_E;           // 4096*2048

    // 1) projections (f16 WMMA, f32 accumulate)
    wmma_gemm_kernel<HID_E><<<dim3(HID_E / 64, SEQ_L / 64), 128, 0, stream>>>(
        hs, nullptr, Wq, qb, HID_E);
    wmma_gemm_kernel<KVW><<<dim3(KVW / 64, SEQ_L / 64), 128, 0, stream>>>(
        hs, nullptr, Wk, kb, HID_E);
    wmma_gemm_kernel<KVW><<<dim3(KVW / 64, SEQ_L / 64), 128, 0, stream>>>(
        hs, nullptr, Wv, vb, HID_E);

    // 2) gate = logsigmoid(k_flat)/16  (pool is identity at these dims)
    {
        int n = SEQ_L * KVW;
        gate_kernel<<<(n + 255) / 256, 256, 0, stream>>>(kb, glb, n);
    }

    // 3) softmax branch: RoPE + sliding-window attention (reads raw q,k,v)
    swa_kernel<<<dim3(NHEAD, NCHUNK), 32, 0, stream>>>(qb, kb, vb, yb);

    // 4) linear branch feature maps: in-place row softmax on q and k
    softmax64_kernel<<<(SEQ_L * NHEAD) / 8, 256, 0, stream>>>(qb, SEQ_L * NHEAD);
    softmax64_kernel<<<(SEQ_L * NKVH) / 8, 256, 0, stream>>>(kb, SEQ_L * NKVH);

    // 5) GLA recurrent scan
    gla_kernel<<<NHEAD, 64, 0, stream>>>(qb, kb, glb, vb, olib);

    // 6) out = (0.5*y + 0.5*o_li) @ Wo
    wmma_gemm_kernel<HID_E><<<dim3(HID_E / 64, SEQ_L / 64), 128, 0, stream>>>(
        yb, olib, Wo, out, HID_E);
}
